// Enet_81037442941606
// MI455X (gfx1250) — compile-verified, matrix operand order checked by reading, not run on hardware
//
#include <hip/hip_runtime.h>
#include <hip/hip_bf16.h>

// ---------------------------------------------------------------------------
// Problem constants (match reference)
// ---------------------------------------------------------------------------
#define B_    32
#define S_    128
#define D_    1024
#define C_    65536
#define N_    512
#define NNZ_  262144
#define EPS_  1e-5f

typedef __attribute__((ext_vector_type(16))) __bf16 v16bf;
typedef __attribute__((ext_vector_type(8)))  float  v8f;

union Frag { v16bf v; unsigned u[8]; };

// Single v_perm_b32: low16 = a[31:16], high16 = b[31:16]  (truncating bf16 pack)
__device__ __forceinline__ unsigned pack_bf16(float a, float b) {
  return __builtin_amdgcn_perm(__float_as_uint(b), __float_as_uint(a), 0x07060302u);
}

// ---------------------------------------------------------------------------
// 1) BatchNorm statistics over the gathered rows: mean/invstd per feature d.
// ---------------------------------------------------------------------------
__global__ __launch_bounds__(256) void bn_stats(const float* __restrict__ enc,
                                                const int* __restrict__ bidx,
                                                const int* __restrict__ tgt,
                                                float* __restrict__ mean,
                                                float* __restrict__ invstd) {
  __shared__ int sb[N_], st[N_];
  const int tid = threadIdx.x;
  for (int i = tid; i < N_; i += 256) { sb[i] = bidx[i]; st[i] = tgt[i]; }
  __syncthreads();
  const int d = blockIdx.x * 256 + tid;
  float s = 0.f, ss = 0.f;
  for (int n = 0; n < N_; ++n) {
    float x = enc[((size_t)sb[n] * S_ + st[n]) * D_ + d];
    s += x; ss += x * x;
  }
  const float m = s * (1.f / N_);
  const float var = ss * (1.f / N_) - m * m;
  mean[d] = m;
  invstd[d] = rsqrtf(var + EPS_);
}

// ---------------------------------------------------------------------------
// 2) Build normalized A-matrix h0 in bf16 (RNE), row-major [N_, D_].
// ---------------------------------------------------------------------------
__global__ __launch_bounds__(256) void build_h0(const float* __restrict__ enc,
                                                const int* __restrict__ bidx,
                                                const int* __restrict__ tgt,
                                                const float* __restrict__ mean,
                                                const float* __restrict__ invstd,
                                                unsigned short* __restrict__ h0) {
  const int i = blockIdx.x * 256 + threadIdx.x;      // < N_*D_
  const int n = i >> 10;
  const int d = i & (D_ - 1);
  float x = enc[((size_t)bidx[n] * S_ + tgt[n]) * D_ + d];
  float h = (x - mean[d]) * invstd[d];
  unsigned u = __float_as_uint(h);
  h0[i] = (unsigned short)((u + 0x7FFFu + ((u >> 16) & 1u)) >> 16);
}

// ---------------------------------------------------------------------------
// 3) GEMM + bias + swish with WMMA bf16, writing h1 transposed: h1T[C_, N_].
//    Block tile 128x128, BK=32; 8 waves, each wave 64x32 (4x2 WMMA tiles).
//    Double-buffered LDS, one barrier per K-iteration.
// ---------------------------------------------------------------------------
#define BM 128
#define BN 128
#define BK 32
#define KT_ (D_ / BK)
#define LDT 17   // dwords per LDS tile row (16 data + 1 pad; 17 coprime 64 banks)

__global__ __launch_bounds__(256) void gemm_swish(const unsigned* __restrict__ h0u,
                                                  const float* __restrict__ w,
                                                  const float* __restrict__ bias,
                                                  float* __restrict__ h1T) {
  __shared__ union {
    struct { unsigned a[2][BM * LDT]; unsigned b[2][BN * LDT]; } t;
    float trans[128 * 132];                       // 128x128 f32 tile, padded
  } sm;

  const int tid  = threadIdx.x;
  const int m0   = (blockIdx.x & 3) * BM;         // 4 row blocks (N_=512)
  const int c0   = (blockIdx.x >> 2) * BN;        // 512 col blocks (C_=65536)

  const int lane = tid & 31;
  const int wave = tid >> 5;
  const int wm   = wave & 1;                      // 64-row slab
  const int wn   = wave >> 1;                     // 32-col slab
  const int r    = lane & 15;
  const int hi   = lane >> 4;

  v8f acc[4][2];
#pragma unroll
  for (int mi = 0; mi < 4; ++mi)
#pragma unroll
    for (int ni = 0; ni < 2; ++ni)
      acc[mi][ni] = (v8f){0.f, 0.f, 0.f, 0.f, 0.f, 0.f, 0.f, 0.f};

  // Global staging assignments (row = tid/2, each thread covers a half row)
  const int row  = tid >> 1;                      // 0..127
  const int part = tid & 1;
  const unsigned* ag = h0u + (size_t)(m0 + row) * (D_ / 2) + part * 8;
  const float*    bg = w   + (size_t)(c0 + row) * D_ + part * 16;

  unsigned* const asw = &sm.t.a[0][0] + row * LDT + part * 8;
  unsigned* const bsw = &sm.t.b[0][0] + row * LDT + part * 8;
  const int abufstride = BM * LDT;
  const int bbufstride = BN * LDT;

  uint4  ra0, ra1;
  float4 rb0, rb1, rb2, rb3;

  // ---- preload k-tile 0 ----
  {
    const uint4*  pa = (const uint4*)ag;
    ra0 = pa[0]; ra1 = pa[1];
    const float4* pb = (const float4*)bg;
    rb0 = pb[0]; rb1 = pb[1]; rb2 = pb[2]; rb3 = pb[3];
  }
  {
    unsigned* as = asw;
    as[0] = ra0.x; as[1] = ra0.y; as[2] = ra0.z; as[3] = ra0.w;
    as[4] = ra1.x; as[5] = ra1.y; as[6] = ra1.z; as[7] = ra1.w;
    unsigned* bs = bsw;
    bs[0] = pack_bf16(rb0.x, rb0.y); bs[1] = pack_bf16(rb0.z, rb0.w);
    bs[2] = pack_bf16(rb1.x, rb1.y); bs[3] = pack_bf16(rb1.z, rb1.w);
    bs[4] = pack_bf16(rb2.x, rb2.y); bs[5] = pack_bf16(rb2.z, rb2.w);
    bs[6] = pack_bf16(rb3.x, rb3.y); bs[7] = pack_bf16(rb3.z, rb3.w);
  }
  __syncthreads();

#pragma unroll 2
  for (int kt = 0; kt < KT_; ++kt) {
    const int cur = kt & 1;

    // ---- issue next tile's global loads (latency overlapped with WMMA) ----
    if (kt + 1 < KT_) {
      const int knext = (kt + 1) * BK;
      const uint4*  pa = (const uint4*)(ag + (knext >> 1));
      ra0 = pa[0]; ra1 = pa[1];
      const float4* pb = (const float4*)(bg + knext);
      rb0 = pb[0]; rb1 = pb[1]; rb2 = pb[2]; rb3 = pb[3];
    }

    // ---- fragments per ISA 16-bit 16x32 layout ----
    const unsigned* abuf = &sm.t.a[0][0] + cur * abufstride;
    const unsigned* bbuf = &sm.t.b[0][0] + cur * bbufstride;
    Frag af[4], bf[2];
#pragma unroll
    for (int mi = 0; mi < 4; ++mi) {
      const unsigned* src = abuf + (wm * 64 + mi * 16 + r) * LDT;
#pragma unroll
      for (int p = 0; p < 8; ++p)
        af[mi].u[p] = src[(p < 4 ? 0 : 8) + hi * 4 + (p & 3)];
    }
#pragma unroll
    for (int ni = 0; ni < 2; ++ni) {
      const unsigned* src = bbuf + (wn * 32 + ni * 16 + r) * LDT;
#pragma unroll
      for (int p = 0; p < 8; ++p)
        bf[ni].u[p] = src[(p < 4 ? 0 : 8) + hi * 4 + (p & 3)];
    }
#pragma unroll
    for (int mi = 0; mi < 4; ++mi)
#pragma unroll
      for (int ni = 0; ni < 2; ++ni)
        acc[mi][ni] = __builtin_amdgcn_wmma_f32_16x16x32_bf16(
            false, af[mi].v, false, bf[ni].v, (short)0, acc[mi][ni],
            false, false);

    // ---- stage next tile into the alternate buffer ----
    if (kt + 1 < KT_) {
      const int nxt = cur ^ 1;
      unsigned* as = asw + nxt * abufstride;
      as[0] = ra0.x; as[1] = ra0.y; as[2] = ra0.z; as[3] = ra0.w;
      as[4] = ra1.x; as[5] = ra1.y; as[6] = ra1.z; as[7] = ra1.w;
      unsigned* bs = bsw + nxt * bbufstride;
      bs[0] = pack_bf16(rb0.x, rb0.y); bs[1] = pack_bf16(rb0.z, rb0.w);
      bs[2] = pack_bf16(rb1.x, rb1.y); bs[3] = pack_bf16(rb1.z, rb1.w);
      bs[4] = pack_bf16(rb2.x, rb2.y); bs[5] = pack_bf16(rb2.z, rb2.w);
      bs[6] = pack_bf16(rb3.x, rb3.y); bs[7] = pack_bf16(rb3.z, rb3.w);
      __syncthreads();
    }
  }

  __syncthreads();   // tiles dead; reuse LDS as transpose buffer

  // ---- epilogue: bias + swish (v_exp + v_rcp), stage transposed tile ----
#pragma unroll
  for (int ni = 0; ni < 2; ++ni) {
    const int c_l = wn * 32 + ni * 16 + r;
    const float bv = bias[c0 + c_l];
#pragma unroll
    for (int mi = 0; mi < 4; ++mi) {
#pragma unroll
      for (int v = 0; v < 8; ++v) {
        const int m_l = wm * 64 + mi * 16 + v + hi * 8;
        float x = acc[mi][ni][v] + bv;
        float h = x * __builtin_amdgcn_rcpf(1.f + __expf(-x));  // swish
        sm.trans[c_l * 132 + m_l] = h;
      }
    }
  }
  __syncthreads();

  // ---- coalesced store of h1T[c, m] ----
  for (int i = tid; i < 128 * 128; i += 256) {
    const int c = i >> 7;
    const int m = i & 127;
    h1T[(size_t)(c0 + c) * N_ + (m0 + m)] = sm.trans[c * 132 + m];
  }
}

// ---------------------------------------------------------------------------
// 4) zero the neighbor accumulator nbrT[C_, N_]
// ---------------------------------------------------------------------------
__global__ __launch_bounds__(256) void zero_ws(float* __restrict__ p) {
  const size_t i = ((size_t)blockIdx.x * 256 + threadIdx.x) * 4;
  float4 z = {0.f, 0.f, 0.f, 0.f};
  *(float4*)(p + i) = z;
}

// ---------------------------------------------------------------------------
// 5) SpMM scatter: nbrT[row, :] += val * h1T[col, :].  One wave per edge;
//    contiguous 2KB row per edge -> coalesced loads, hardware f32 atomics.
// ---------------------------------------------------------------------------
__global__ __launch_bounds__(256) void spmm(const int* __restrict__ rows,
                                            const int* __restrict__ cols,
                                            const float* __restrict__ vals,
                                            const float* __restrict__ h1T,
                                            float* __restrict__ nbrT) {
  const int e    = blockIdx.x * 8 + (threadIdx.x >> 5);
  const int lane = threadIdx.x & 31;
  const int rw = rows[e];
  const int cl = cols[e];
  const float v = vals[e];
  const float4* src = (const float4*)(h1T + (size_t)cl * N_);
  float* dst = nbrT + (size_t)rw * N_;
#pragma unroll
  for (int wv = 0; wv < 4; ++wv) {
    const int i4 = wv * 32 + lane;
    float4 h = src[i4];
    unsafeAtomicAdd(dst + i4 * 4 + 0, v * h.x);
    unsafeAtomicAdd(dst + i4 * 4 + 1, v * h.y);
    unsafeAtomicAdd(dst + i4 * 4 + 2, v * h.z);
    unsafeAtomicAdd(dst + i4 * 4 + 3, v * h.w);
  }
}

// ---------------------------------------------------------------------------
// 6) merge + transpose: out[n, c] = h1T[c, n] + nbrT[c, n]
// ---------------------------------------------------------------------------
__global__ __launch_bounds__(256) void merge_t(const float* __restrict__ h1T,
                                               const float* __restrict__ nbrT,
                                               float* __restrict__ out) {
  __shared__ float t[32][33];
  const int bx = blockIdx.x & (C_ / 32 - 1);      // 2048 column tiles
  const int by = blockIdx.x >> 11;                // 16 row tiles
  const int c0 = bx * 32, n0 = by * 32;
  const int tx = threadIdx.x & 31, ty = threadIdx.x >> 5;   // 32 x 8
#pragma unroll
  for (int i = 0; i < 4; ++i) {
    const int c = ty + i * 8;
    const size_t off = (size_t)(c0 + c) * N_ + n0 + tx;
    t[c][tx] = h1T[off] + nbrT[off];
  }
  __syncthreads();
#pragma unroll
  for (int i = 0; i < 4; ++i) {
    const int n = ty + i * 8;
    out[(size_t)(n0 + n) * C_ + c0 + tx] = t[tx][n];
  }
}

// ---------------------------------------------------------------------------
// Launch
// ---------------------------------------------------------------------------
extern "C" void kernel_launch(void* const* d_in, const int* in_sizes, int n_in,
                              void* d_out, int out_size, void* d_ws, size_t ws_size,
                              hipStream_t stream) {
  const float* enc   = (const float*)d_in[0];   // [B,S,D]
  const float* w     = (const float*)d_in[1];   // [C,D]
  const float* bias  = (const float*)d_in[2];   // [C]
  const float* avals = (const float*)d_in[3];   // [NNZ]
  const int*   bidx  = (const int*)d_in[4];     // [N]
  const int*   tgt   = (const int*)d_in[5];     // [N]
  const int*   aidx  = (const int*)d_in[6];     // [2,NNZ]
  float* out = (float*)d_out;

  char* ws = (char*)d_ws;
  float*          mean   = (float*)ws;                                    // 4 KB
  float*          invstd = (float*)(ws + 4096);                           // 4 KB
  unsigned short* h0     = (unsigned short*)(ws + 8192);                  // 1 MB
  float*          h1T    = (float*)(ws + ((size_t)2 << 20));              // 128 MB
  float*          nbrT   = (float*)(ws + ((size_t)2 << 20) + ((size_t)C_ * N_ * 4));

  bn_stats <<<D_ / 256,            256, 0, stream>>>(enc, bidx, tgt, mean, invstd);
  build_h0 <<<(N_ * D_) / 256,     256, 0, stream>>>(enc, bidx, tgt, mean, invstd, h0);
  gemm_swish<<<4 * (C_ / BN),      256, 0, stream>>>((const unsigned*)h0, w, bias, h1T);
  zero_ws  <<<(C_ * N_) / (256*4), 256, 0, stream>>>(nbrT);
  spmm     <<<NNZ_ / 8,            256, 0, stream>>>(aidx, aidx + NNZ_, avals, h1T, nbrT);
  merge_t  <<<(C_/32) * (N_/32),   256, 0, stream>>>(h1T, nbrT, out);
}
